// PhaseAwareAttention_51539607552830
// MI455X (gfx1250) — compile-verified, matrix operand order checked by reading, not run on hardware
//
#include <hip/hip_runtime.h>

// ---------------------------------------------------------------------------
// PhaseAwareAttention on MI455X (gfx1250, wave32, WMMA 16x16x32 f16->f32)
// B=4 S=1024 E=512 H=8 D=64 NB=5
// ---------------------------------------------------------------------------

typedef __attribute__((ext_vector_type(16))) _Float16 v16h;
typedef __attribute__((ext_vector_type(8)))  _Float16 v8h;
typedef __attribute__((ext_vector_type(8)))  float    v8f;

static constexpr int Bc  = 4;
static constexpr int Sc  = 1024;
static constexpr int Ec  = 512;
static constexpr int Hc  = 8;
static constexpr int Dc  = 64;
static constexpr int NBc = 5;

// ---------------------------------------------------------------------------
// Fragment helpers (wave32, 16x16x32 f16).
// A-matrix layout (ISA 7.12.2): lane&15 = row M; lane>=16 selects K half:
//   halves j=0..7  -> K = koff + j        (koff = (lane>>4)*8)
//   halves j=8..15 -> K = koff + 16 + (j-8)
// B is fed as B^T rows (column-major B) with the mirrored pattern (lane&15=N).
// C/D: col N = lane&15, row M = r + 8*(lane>=16), r = VGPR index 0..7.
// ---------------------------------------------------------------------------
__device__ __forceinline__ v8h load8h(const _Float16* p) {
  return *reinterpret_cast<const v8h*>(p);
}

__device__ __forceinline__ v16h frag_rowmajor(const _Float16* base, int ld, int lane) {
  const int r    = lane & 15;
  const int koff = (lane >> 4) << 3;  // 0 or 8
  const _Float16* p = base + r * ld + koff;
  v8h lo = load8h(p);
  v8h hi = load8h(p + 16);
  v16h f;
#pragma unroll
  for (int j = 0; j < 8; ++j) { f[j] = lo[j]; f[j + 8] = hi[j]; }
  return f;
}

__device__ __forceinline__ void load8f(const float* p, float* o) {
  float4 a = *reinterpret_cast<const float4*>(p);
  float4 b = *reinterpret_cast<const float4*>(p + 4);
  o[0] = a.x; o[1] = a.y; o[2] = a.z; o[3] = a.w;
  o[4] = b.x; o[5] = b.y; o[6] = b.z; o[7] = b.w;
}

// ---------------------------------------------------------------------------
// fp32 -> fp16 convert
// ---------------------------------------------------------------------------
__global__ void cvt_f16_kernel(const float* __restrict__ src,
                               _Float16* __restrict__ dst, int n) {
  int i = blockIdx.x * blockDim.x + threadIdx.x;
  if (i < n) dst[i] = (_Float16)src[i];
}

// ---------------------------------------------------------------------------
// xf = avg_pool1d(x, k=15, pad=7, count_include_pad)  over S, layout (B,S,E)
// ---------------------------------------------------------------------------
__global__ void pool_kernel(const float* __restrict__ x, float* __restrict__ xf) {
  int idx = blockIdx.x * blockDim.x + threadIdx.x;  // B*S*E = 2^21
  if (idx >= Bc * Sc * Ec) return;
  int e = idx & (Ec - 1);
  int s = (idx >> 9) & (Sc - 1);
  int b = idx >> 19;
  float acc = 0.f;
#pragma unroll
  for (int k = -7; k <= 7; ++k) {
    int s2 = s + k;
    if (s2 >= 0 && s2 < Sc) acc += x[((b << 10) + s2) * Ec + e];
  }
  xf[idx] = acc * (1.0f / 15.0f);
}

// ---------------------------------------------------------------------------
// per band: h = tanh(dwconv15(xf) + hb1), layout (B,S,E)
// ---------------------------------------------------------------------------
__global__ void band_h_kernel(const float* __restrict__ xf,
                              const float* __restrict__ hw1,
                              const float* __restrict__ hb1,
                              float* __restrict__ hbuf, int band) {
  int idx = blockIdx.x * blockDim.x + threadIdx.x;
  if (idx >= Bc * Sc * Ec) return;
  int e = idx & (Ec - 1);
  int s = (idx >> 9) & (Sc - 1);
  int b = idx >> 19;
  const float* w = hw1 + (band * Ec + e) * 15;
  float acc = hb1[band * Ec + e];
#pragma unroll
  for (int k = 0; k < 15; ++k) {
    int s2 = s + k - 7;
    if (s2 >= 0 && s2 < Sc) acc += xf[((b << 10) + s2) * Ec + e] * w[k];
  }
  hbuf[idx] = tanhf(acc);
}

// ---------------------------------------------------------------------------
// per band: hilb = dwconv7(h)+hb2 ; phase=atan2(hilb,xf) ;
// pf = [mean cos, mean sin] over E ; gate MLP -> pg[band,b,s,h]
// one block per (b,s)
// ---------------------------------------------------------------------------
__global__ void band_gate_kernel(const float* __restrict__ xf,
                                 const float* __restrict__ hbuf,
                                 const float* __restrict__ hw2,
                                 const float* __restrict__ hb2,
                                 const float* __restrict__ pw1,
                                 const float* __restrict__ pb1,
                                 const float* __restrict__ pw2,
                                 const float* __restrict__ pb2,
                                 float* __restrict__ pg, int band) {
  int bs = blockIdx.x;          // 0 .. B*S-1
  int b = bs >> 10;
  int s = bs & (Sc - 1);
  int tid = threadIdx.x;
  __shared__ float scos[256], ssin[256];
  float csum = 0.f, ssum = 0.f;
  for (int e = tid; e < Ec; e += 256) {
    const float* w = hw2 + (band * Ec + e) * 7;
    float acc = hb2[band * Ec + e];
#pragma unroll
    for (int k = 0; k < 7; ++k) {
      int s2 = s + k - 3;
      if (s2 >= 0 && s2 < Sc) acc += hbuf[((b << 10) + s2) * Ec + e] * w[k];
    }
    float ph = atan2f(acc, xf[((b << 10) + s) * Ec + e]);
    csum += cosf(ph);
    ssum += sinf(ph);
  }
  scos[tid] = csum; ssin[tid] = ssum;
  __syncthreads();
  for (int off = 128; off > 0; off >>= 1) {
    if (tid < off) { scos[tid] += scos[tid + off]; ssin[tid] += ssin[tid + off]; }
    __syncthreads();
  }
  if (tid == 0) {
    float pf0 = scos[0] * (1.0f / Ec);
    float pf1 = ssin[0] * (1.0f / Ec);
    float t[Hc];
#pragma unroll
    for (int hh = 0; hh < Hc; ++hh)
      t[hh] = tanhf(pf0 * pw1[(band * Hc + hh) * 2 + 0] +
                    pf1 * pw1[(band * Hc + hh) * 2 + 1] + pb1[band * Hc + hh]);
#pragma unroll
    for (int hh = 0; hh < Hc; ++hh) {
      float acc = pb2[band * Hc + hh];
#pragma unroll
      for (int j = 0; j < Hc; ++j) acc += t[j] * pw2[(band * Hc + hh) * Hc + j];
      pg[(((size_t)band * Bc + b) * Sc + s) * Hc + hh] = 1.f / (1.f + __expf(-acc));
    }
  }
}

// ---------------------------------------------------------------------------
// WMMA projection: Y = A(4096x512,f16) @ W^T (W: 512x512,f16) + bias
// mode 0: f16 heads layout (B,H,S,D)
// mode 1: f32 row-major (M,N)
// mode 2: f16 transposed heads layout (B,H,D,S)   (for V -> AV B-fragments)
// one wave per 16x16 tile.
// ---------------------------------------------------------------------------
__global__ void gemm_proj_kernel(const _Float16* __restrict__ A,
                                 const _Float16* __restrict__ W,
                                 const float* __restrict__ bias,
                                 _Float16* __restrict__ out_h,
                                 float* __restrict__ out_f, int mode) {
  constexpr int M = Bc * Sc, N = Ec, K = Ec;
  int wave = (blockIdx.x * blockDim.x + threadIdx.x) >> 5;
  int lane = threadIdx.x & 31;
  constexpr int tilesN = N >> 4;
  if (wave >= (M >> 4) * tilesN) return;  // wave-uniform
  int tileM = wave / tilesN;
  int tileN = wave - tileM * tilesN;
  const _Float16* Abase = A + (size_t)tileM * 16 * K;
  const _Float16* Wbase = W + (size_t)tileN * 16 * K;
  v8f c = {0.f, 0.f, 0.f, 0.f, 0.f, 0.f, 0.f, 0.f};
  for (int kk = 0; kk < K; kk += 32) {
    if (kk + 32 < K) __builtin_prefetch(Abase + kk + 32, 0, 0);
    v16h a = frag_rowmajor(Abase + kk, K, lane);
    v16h bfr = frag_rowmajor(Wbase + kk, K, lane);
    c = __builtin_amdgcn_wmma_f32_16x16x32_f16(false, a, false, bfr,
                                               (short)0, c, false, false);
  }
  int n = tileN * 16 + (lane & 15);
  int hi = (lane >> 4) & 1;
  float bv = bias ? bias[n] : 0.f;
  int h = n >> 6, d = n & (Dc - 1);
#pragma unroll
  for (int r = 0; r < 8; ++r) {
    int m = tileM * 16 + r + hi * 8;
    int b = m >> 10, s = m & (Sc - 1);
    float v = c[r] + bv;
    if (mode == 0) {
      out_h[((((size_t)b * Hc + h) << 10) + s) * Dc + d] = (_Float16)v;
    } else if (mode == 1) {
      out_f[(size_t)m * N + n] = v;
    } else {
      out_h[((((size_t)b * Hc + h) * Dc + d) << 10) + s] = (_Float16)v;
    }
  }
}

// ---------------------------------------------------------------------------
// scores[b,h,q,k] = (Q . K)/sqrt(D), per (b,h) 1024x1024, Kdim=64
// ---------------------------------------------------------------------------
__global__ void scores_kernel(const _Float16* __restrict__ Qh,
                              const _Float16* __restrict__ Kh,
                              float* __restrict__ scores) {
  int wave = (blockIdx.x * blockDim.x + threadIdx.x) >> 5;
  int lane = threadIdx.x & 31;
  if (wave >= Bc * Hc * 64 * 64) return;  // wave-uniform
  int bh = wave >> 12;
  int t  = wave & 4095;
  int tq = t >> 6, tk = t & 63;
  const _Float16* Qb = Qh + ((size_t)bh << 10) * Dc + (size_t)tq * 16 * Dc;
  const _Float16* Kb = Kh + ((size_t)bh << 10) * Dc + (size_t)tk * 16 * Dc;
  v8f c = {0.f, 0.f, 0.f, 0.f, 0.f, 0.f, 0.f, 0.f};
#pragma unroll
  for (int kk = 0; kk < Dc; kk += 32) {
    v16h a = frag_rowmajor(Qb + kk, Dc, lane);
    v16h bfr = frag_rowmajor(Kb + kk, Dc, lane);
    c = __builtin_amdgcn_wmma_f32_16x16x32_f16(false, a, false, bfr,
                                               (short)0, c, false, false);
  }
  int n = tk * 16 + (lane & 15);
  int hi = (lane >> 4) & 1;
  float* sb = scores + (size_t)bh * Sc * Sc;
#pragma unroll
  for (int r = 0; r < 8; ++r) {
    int m = tq * 16 + r + hi * 8;
    sb[(size_t)m * Sc + n] = c[r] * 0.125f;  // 1/sqrt(64)
  }
}

// ---------------------------------------------------------------------------
// per (band,b,h,q) row: rowmax = max(s*gate), rowscale = mix[band]/sum(exp)
// one block per row; scores are L2-resident (128MB < 192MB L2).
// ---------------------------------------------------------------------------
__global__ void stats_kernel(const float* __restrict__ scores,
                             const float* __restrict__ pg,
                             const float* __restrict__ phase_scaling,
                             float* __restrict__ rowmax,
                             float* __restrict__ rowscale) {
  int row  = blockIdx.x;                 // NB*B*H*S rows
  int band = row >> 15;                  // / 32768
  int rem  = row & 32767;
  int b = rem >> 13;
  int h = (rem >> 10) & (Hc - 1);
  int q = rem & (Sc - 1);
  float mod = pg[(((size_t)band * Bc + b) * Sc + q) * Hc + h];
  const float* srow = scores + (((size_t)(b * Hc + h) << 10) + q) * Sc;
  int tid = threadIdx.x;
  __shared__ float sm[256];
  float mx = -3.0e38f;
  for (int k = tid; k < Sc; k += 256) mx = fmaxf(mx, srow[k] * mod);
  sm[tid] = mx;
  __syncthreads();
  for (int off = 128; off > 0; off >>= 1) {
    if (tid < off) sm[tid] = fmaxf(sm[tid], sm[tid + off]);
    __syncthreads();
  }
  float rmax = sm[0];
  __syncthreads();
  float su = 0.f;
  for (int k = tid; k < Sc; k += 256) su += __expf(srow[k] * mod - rmax);
  sm[tid] = su;
  __syncthreads();
  for (int off = 128; off > 0; off >>= 1) {
    if (tid < off) sm[tid] += sm[tid + off];
    __syncthreads();
  }
  if (tid == 0) {
    float pm = -3.0e38f;
#pragma unroll
    for (int i = 0; i < NBc; ++i) pm = fmaxf(pm, phase_scaling[i]);
    float den = 0.f;
#pragma unroll
    for (int i = 0; i < NBc; ++i) den += __expf(phase_scaling[i] - pm);
    float mix = __expf(phase_scaling[band] - pm) / den;
    rowmax[row]   = rmax;
    rowscale[row] = mix / sm[0];
  }
}

// ---------------------------------------------------------------------------
// Fused softmax(scores*gate) @ V summed over all 5 bands.
// One wave owns a 16x64 output slab (qtile x full head-dim):
//  - each score element is loaded once and exp'd once per band (kk outer,
//    band inner) -> minimal TRANS work;
//  - V comes from the pre-transposed Vt (B,H,D,S) so B-fragments are two
//    global_load_b128 per tile, loaded once per kk and reused by all bands;
//  - 20 WMMAs per kk iteration amortize 12 vector loads.
// Output: combined (B,S,E) f16 (feeds the Wo projection GEMM).
// ---------------------------------------------------------------------------
__global__ void av_kernel(const float* __restrict__ scores,
                          const _Float16* __restrict__ Vt,
                          const float* __restrict__ pg,
                          const float* __restrict__ rowmax,
                          const float* __restrict__ rowscale,
                          _Float16* __restrict__ comb) {
  int wave = (blockIdx.x * blockDim.x + threadIdx.x) >> 5;
  int lane = threadIdx.x & 31;
  if (wave >= Bc * Hc * (Sc / 16)) return;  // wave-uniform
  int qt = wave & 63;
  int bh = wave >> 6;
  int b = bh >> 3, h = bh & (Hc - 1);
  int r15 = lane & 15;
  int hi  = (lane >> 4) & 1;
  int q = qt * 16 + r15;                          // A row owned by this lane
  const float* srow = scores + (((size_t)bh << 10) + q) * Sc;
  const _Float16* vbase = Vt + (((size_t)bh * Dc) << 10);

  // per-lane, per-band softmax constants for this query row
  float amax[NBc], ascale[NBc], amod[NBc];
#pragma unroll
  for (int band = 0; band < NBc; ++band) {
    size_t ridx = ((size_t)band << 15) + ((size_t)bh << 10) + q;
    amax[band]   = rowmax[ridx];
    ascale[band] = rowscale[ridx];
    amod[band]   = pg[(((size_t)band * Bc + b) * Sc + q) * Hc + h];
  }

  v8f c0 = {0.f, 0.f, 0.f, 0.f, 0.f, 0.f, 0.f, 0.f};
  v8f c1 = c0, c2 = c0, c3 = c0;

  for (int kk = 0; kk < Sc; kk += 32) {
    // scores for this lane's query row, WMMA A-element order
    float sv[16];
    load8f(srow + kk + (hi << 3), sv);           // K = kk+koff   .. +7
    load8f(srow + kk + 16 + (hi << 3), sv + 8);  // K = kk+16+koff.. +7
    // V fragments (B^T rows from Vt), shared across all bands
    v16h b0 = frag_rowmajor(vbase + ((size_t)0  << 10) + kk, Sc, lane);
    v16h b1 = frag_rowmajor(vbase + ((size_t)16 << 10) + kk, Sc, lane);
    v16h b2 = frag_rowmajor(vbase + ((size_t)32 << 10) + kk, Sc, lane);
    v16h b3 = frag_rowmajor(vbase + ((size_t)48 << 10) + kk, Sc, lane);
#pragma unroll
    for (int band = 0; band < NBc; ++band) {
      const float m = amax[band], sc = ascale[band], md = amod[band];
      v16h a;
#pragma unroll
      for (int j = 0; j < 16; ++j)
        a[j] = (_Float16)(__expf(sv[j] * md - m) * sc);
      c0 = __builtin_amdgcn_wmma_f32_16x16x32_f16(false, a, false, b0,
                                                  (short)0, c0, false, false);
      c1 = __builtin_amdgcn_wmma_f32_16x16x32_f16(false, a, false, b1,
                                                  (short)0, c1, false, false);
      c2 = __builtin_amdgcn_wmma_f32_16x16x32_f16(false, a, false, b2,
                                                  (short)0, c2, false, false);
      c3 = __builtin_amdgcn_wmma_f32_16x16x32_f16(false, a, false, b3,
                                                  (short)0, c3, false, false);
    }
  }

  // store combined (B,S,E): row m = qt*16 + r + 8*hi, col e = h*64 + dt*16 + r15
#pragma unroll
  for (int r = 0; r < 8; ++r) {
    int m = qt * 16 + r + hi * 8;
    size_t rowoff = (((size_t)b << 10) + m) * Ec + h * Dc + r15;
    comb[rowoff +  0] = (_Float16)c0[r];
    comb[rowoff + 16] = (_Float16)c1[r];
    comb[rowoff + 32] = (_Float16)c2[r];
    comb[rowoff + 48] = (_Float16)c3[r];
  }
}

// ---------------------------------------------------------------------------
// y = proj + x ; LayerNorm(y)*g + b  -> d_out. One block per (b,s).
// ---------------------------------------------------------------------------
__global__ void ln_kernel(const float* __restrict__ proj,
                          const float* __restrict__ x,
                          const float* __restrict__ g,
                          const float* __restrict__ bt,
                          float* __restrict__ out) {
  int bs = blockIdx.x;
  const float* pr = proj + (size_t)bs * Ec;
  const float* xr = x + (size_t)bs * Ec;
  float* orow = out + (size_t)bs * Ec;
  int tid = threadIdx.x;
  __shared__ float s1[256], s2[256];
  float y0 = pr[tid] + xr[tid];
  float y1 = pr[tid + 256] + xr[tid + 256];
  s1[tid] = y0 + y1;
  s2[tid] = y0 * y0 + y1 * y1;
  __syncthreads();
  for (int off = 128; off > 0; off >>= 1) {
    if (tid < off) { s1[tid] += s1[tid + off]; s2[tid] += s2[tid + off]; }
    __syncthreads();
  }
  float mu  = s1[0] * (1.0f / Ec);
  float var = s2[0] * (1.0f / Ec) - mu * mu;
  float inv = rsqrtf(var + 1e-5f);
  orow[tid]       = (y0 - mu) * inv * g[tid] + bt[tid];
  orow[tid + 256] = (y1 - mu) * inv * g[tid + 256] + bt[tid + 256];
}

// ---------------------------------------------------------------------------
// host side
// ---------------------------------------------------------------------------
extern "C" void kernel_launch(void* const* d_in, const int* in_sizes, int n_in,
                              void* d_out, int out_size, void* d_ws, size_t ws_size,
                              hipStream_t stream) {
  (void)in_sizes; (void)n_in; (void)out_size; (void)ws_size;
  const float* x   = (const float*)d_in[0];
  const float* hw1 = (const float*)d_in[1];
  const float* hb1 = (const float*)d_in[2];
  const float* hw2 = (const float*)d_in[3];
  const float* hb2 = (const float*)d_in[4];
  const float* pw1 = (const float*)d_in[5];
  const float* pb1 = (const float*)d_in[6];
  const float* pw2 = (const float*)d_in[7];
  const float* pb2 = (const float*)d_in[8];
  const float* Wq  = (const float*)d_in[9];
  const float* bq  = (const float*)d_in[10];
  const float* Wk  = (const float*)d_in[11];
  const float* bk  = (const float*)d_in[12];
  const float* Wv  = (const float*)d_in[13];
  const float* bv  = (const float*)d_in[14];
  const float* Wo  = (const float*)d_in[15];
  const float* bo  = (const float*)d_in[16];
  const float* phase_scaling = (const float*)d_in[17];
  const float* ln_g = (const float*)d_in[18];
  const float* ln_b = (const float*)d_in[19];
  float* out = (float*)d_out;

  // workspace bump allocator (256B aligned)
  char* ws = (char*)d_ws;
  size_t off = 0;
  auto alloc = [&](size_t bytes) -> void* {
    off = (off + 255) & ~(size_t)255;
    void* p = ws + off;
    off += bytes;
    return p;
  };
  const size_t NXE = (size_t)Bc * Sc * Ec;    // 2M elements
  float*    xf      = (float*)alloc(NXE * 4);
  float*    hbuf    = (float*)alloc(NXE * 4);
  float*    pgbuf   = (float*)alloc((size_t)NBc * Bc * Sc * Hc * 4);
  float*    rowmax  = (float*)alloc((size_t)NBc * Bc * Hc * Sc * 4);
  float*    rowscl  = (float*)alloc((size_t)NBc * Bc * Hc * Sc * 4);
  _Float16* xh      = (_Float16*)alloc(NXE * 2);
  _Float16* wqh     = (_Float16*)alloc((size_t)Ec * Ec * 2);
  _Float16* wkh     = (_Float16*)alloc((size_t)Ec * Ec * 2);
  _Float16* wvh     = (_Float16*)alloc((size_t)Ec * Ec * 2);
  _Float16* woh     = (_Float16*)alloc((size_t)Ec * Ec * 2);
  _Float16* Qh      = (_Float16*)alloc(NXE * 2);
  _Float16* Kh      = (_Float16*)alloc(NXE * 2);
  _Float16* Vt      = (_Float16*)alloc(NXE * 2);   // (B,H,D,S)
  _Float16* comb    = (_Float16*)alloc(NXE * 2);
  float*    proj    = (float*)alloc(NXE * 4);
  float*    scores  = (float*)alloc((size_t)Bc * Hc * Sc * Sc * 4);  // 128MB, L2-resident

  // 1) converts
  cvt_f16_kernel<<<(int)((NXE + 255) / 256), 256, 0, stream>>>(x, xh, (int)NXE);
  cvt_f16_kernel<<<(Ec * Ec + 255) / 256, 256, 0, stream>>>(Wq, wqh, Ec * Ec);
  cvt_f16_kernel<<<(Ec * Ec + 255) / 256, 256, 0, stream>>>(Wk, wkh, Ec * Ec);
  cvt_f16_kernel<<<(Ec * Ec + 255) / 256, 256, 0, stream>>>(Wv, wvh, Ec * Ec);
  cvt_f16_kernel<<<(Ec * Ec + 255) / 256, 256, 0, stream>>>(Wo, woh, Ec * Ec);

  // 2) band-pass proxy
  pool_kernel<<<(int)(NXE / 256), 256, 0, stream>>>(x, xf);

  // 3) Q/K/V projections (WMMA); V stored pre-transposed per head
  const int projTiles  = (Bc * Sc / 16) * (Ec / 16);   // 8192
  const int projBlocks = projTiles * 32 / 256;         // 1024
  gemm_proj_kernel<<<projBlocks, 256, 0, stream>>>(xh, wqh, bq, Qh, nullptr, 0);
  gemm_proj_kernel<<<projBlocks, 256, 0, stream>>>(xh, wkh, bk, Kh, nullptr, 0);
  gemm_proj_kernel<<<projBlocks, 256, 0, stream>>>(xh, wvh, bv, Vt, nullptr, 2);

  // 4) scores = QK^T / sqrt(D)  (WMMA), L2-resident fp32
  const int scWaves = Bc * Hc * 64 * 64;               // 131072
  scores_kernel<<<scWaves * 32 / 256, 256, 0, stream>>>(Qh, Kh, scores);

  // 5) per-band depthwise convs + phase gate MLP
  for (int band = 0; band < NBc; ++band) {
    band_h_kernel<<<(int)(NXE / 256), 256, 0, stream>>>(xf, hw1, hb1, hbuf, band);
    band_gate_kernel<<<Bc * Sc, 256, 0, stream>>>(xf, hbuf, hw2, hb2,
                                                  pw1, pb1, pw2, pb2, pgbuf, band);
  }

  // 6) per-(band,row) softmax stats; band-mix softmax folded into rowscale
  stats_kernel<<<NBc * Bc * Hc * Sc, 256, 0, stream>>>(scores, pgbuf, phase_scaling,
                                                       rowmax, rowscl);

  // 7) fused softmax*V over all bands (WMMA) -> combined f16
  const int avWaves = Bc * Hc * (Sc / 16);             // 2048
  av_kernel<<<avWaves * 32 / 256, 256, 0, stream>>>(scores, Vt, pgbuf,
                                                    rowmax, rowscl, comb);

  // 8) output projection (WMMA, fp32 out + bias)
  gemm_proj_kernel<<<projBlocks, 256, 0, stream>>>(comb, woh, bo, nullptr, proj, 1);

  // 9) residual + LayerNorm -> d_out
  ln_kernel<<<Bc * Sc, 256, 0, stream>>>(proj, x, ln_g, ln_b, out);
}
